// StackedFlow_39642548142732
// MI455X (gfx1250) — matrix operand order repaired and checked by reading, not compiled, not run on hardware
//
#include <hip/hip_runtime.h>
#include <hip/hip_bf16.h>

typedef _Float16 v8h  __attribute__((ext_vector_type(8)));
typedef _Float16 v16h __attribute__((ext_vector_type(16)));
typedef float    v8f  __attribute__((ext_vector_type(8)));

#define NLAYER 4
#define BATCH  32768
#define DFEAT  64
#define HDIM   256
#define DMASK  32
#define NSTEP  8
#define EPSV   1e-3f

// strides in halves
#define ST_W0_NVP 40
#define ST_W0_ODE 72
#define ST_W1     264
#define ST_W2     264
#define ST_H2     264
#define ST_SCR_H  144
#define ST_SCR_F  72

// LDS byte offsets
#define OFF_WA   0        // 256*72 halves = 36864 B (nvp uses 256*40)
#define OFF_WB   36864    // 256*264 halves = 135168 B
#define OFF_WC   172032   // 64*264 halves = 33792 B
#define OFF_BIAS 205824   // 1024 floats = 4096 B
#define OFF_SCR  209920   // 8 waves * 16*72 floats = 36864 B
#define OFF_H2   246784   // 8 waves * 16*264 halves = 67584 B
#define LDS_TOTAL 314368

__device__ __forceinline__ void wait_lds() {
    asm volatile("s_wait_dscnt 0x0" ::: "memory");
}

__device__ __forceinline__ v16h cat16(v8h lo, v8h hi) {
    return __builtin_shufflevector(lo, hi, 0,1,2,3,4,5,6,7,8,9,10,11,12,13,14,15);
}

// A-matrix fragment (16xK tile, row-major K-contiguous in LDS)
__device__ __forceinline__ v16h afrag(const _Float16* base, int stride, int kc, int lane) {
    int m = lane & 15, sel = lane >> 4;
    const _Float16* q = base + m * stride + kc + sel * 8;
    return cat16(*(const v8h*)q, *(const v8h*)(q + 16));
}

// B-matrix fragment from transposed weights Wt[n][k]
__device__ __forceinline__ v16h bfrag(const _Float16* Wt, int stride, int nrow, int kc, int lane) {
    int n = lane & 15, sel = lane >> 4;
    const _Float16* q = Wt + (nrow + n) * stride + kc + sel * 8;
    return cat16(*(const v8h*)q, *(const v8h*)(q + 16));
}

__device__ __forceinline__ v8f wmma16(v16h a, v16h b, v8f c) {
    return __builtin_amdgcn_wmma_f32_16x16x32_f16(false, a, false, b, (short)0, c, false, false);
}

// store C/D-layout f32 tile to LDS as f16, row m = r + 8*sel, col = col0 + n16
__device__ __forceinline__ void store_tile_f16(_Float16* base, int stride, int col0, v8f c, int lane) {
    int n = lane & 15, sel = lane >> 4;
    _Float16* p = base + (sel * 8) * stride + col0 + n;
#pragma unroll
    for (int r = 0; r < 8; ++r) p[r * stride] = (_Float16)c[r];
}

__device__ __forceinline__ float act_tanh(float x) {
#if __has_builtin(__builtin_amdgcn_tanhf)
    return __builtin_amdgcn_tanhf(x);
#else
    float e = __expf(2.0f * x);
    return 1.0f - 2.0f / (e + 1.0f);
#endif
}

__device__ __forceinline__ void copy16(void* dst, const void* src, int bytes, int tid) {
    uint4* d = (uint4*)dst; const uint4* s = (const uint4*)src;
    int n = bytes >> 4;
    for (int i = tid; i < n; i += 256) d[i] = s[i];
}

// 3-layer MLP on a 16-row tile. Input (16 x 32*NCH1, f16) must already be in scrH.
// GEMM1: K=32*NCH1 -> 256 (+act) ; GEMM2: 256 -> 256 (+act) ; GEMM3: 256 -> 64.
// GEMM1 and GEMM2 are software-pipelined by one N-tile so that the activation of tile
// nt-1 fills the WMMA->VALU hazard slots of tile nt's WMMA chain.
template<int NCH1, bool TANH>
__device__ __forceinline__ void mlp3(
    _Float16* scrH, _Float16* h2w,
    const _Float16* W0t, const int st0,
    const _Float16* W1t, const _Float16* W2t,
    const float* b0, const float* b1, const float* b2,
    float t, const float* w0r0,
    v8f out[4], int lane)
{
    const int n16 = lane & 15;
    wait_lds();
    v16h a[NCH1];
#pragma unroll
    for (int i = 0; i < NCH1; ++i) a[i] = afrag(scrH, ST_SCR_H, i * 32, lane);

    v16h hA[8];

    auto act8 = [&](v8f& c) {
#pragma unroll
        for (int r = 0; r < 8; ++r)
            c[r] = TANH ? act_tanh(c[r]) : fmaxf(c[r], 0.0f);
    };

    // ---- GEMM1 + act -> hA (register A-frags) via scratch pair-conversion ----
    {
        v8f cp;
#pragma unroll
        for (int nt = 0; nt < 16; ++nt) {
            float bias = b0[nt * 16 + n16];
            if (w0r0 != nullptr) bias += t * w0r0[nt * 16 + n16];
            v8f c;
#pragma unroll
            for (int r = 0; r < 8; ++r) c[r] = bias;
#pragma unroll
            for (int i = 0; i < NCH1; ++i)
                c = wmma16(a[i], bfrag(W0t, st0, nt * 16, i * 32, lane), c);
            if (nt > 0) {
                const int j = nt - 1;
                act8(cp);
                store_tile_f16(scrH, ST_SCR_H, (j & 1) * 16, cp, lane);
                if (j & 1) {           // pair complete -> reload as A-frag
                    wait_lds();
                    hA[j >> 1] = afrag(scrH, ST_SCR_H, 0, lane);
                }
            }
            cp = c;
        }
        act8(cp);
        store_tile_f16(scrH, ST_SCR_H, 16, cp, lane);
        wait_lds();
        hA[7] = afrag(scrH, ST_SCR_H, 0, lane);
    }

    // ---- GEMM2 + act -> h2 (LDS), pipelined by one tile ----
    {
        auto g2tile = [&](int nt) -> v8f {
            float bias = b1[nt * 16 + n16];
            v8f c;
#pragma unroll
            for (int r = 0; r < 8; ++r) c[r] = bias;
#pragma unroll
            for (int kc = 0; kc < 8; ++kc)
                c = wmma16(hA[kc], bfrag(W1t, ST_W1, nt * 16, kc * 32, lane), c);
            return c;
        };
        v8f cq = g2tile(0);
#pragma unroll 1
        for (int nt = 1; nt < 16; ++nt) {
            v8f c = g2tile(nt);
            act8(cq);
            store_tile_f16(h2w, ST_H2, (nt - 1) * 16, cq, lane);
            cq = c;
        }
        act8(cq);
        store_tile_f16(h2w, ST_H2, 15 * 16, cq, lane);
    }
    wait_lds();
#pragma unroll
    for (int kc = 0; kc < 8; ++kc) hA[kc] = afrag(h2w, ST_H2, kc * 32, lane);

    // ---- GEMM3 -> out (f32 C-layout registers) ----
#pragma unroll
    for (int nt = 0; nt < 4; ++nt) {
        float bias = b2[nt * 16 + n16];
        v8f c;
#pragma unroll
        for (int r = 0; r < 8; ++r) c[r] = bias;
#pragma unroll
        for (int kc = 0; kc < 8; ++kc)
            c = wmma16(hA[kc], bfrag(W2t, ST_W2, nt * 16, kc * 32, lane), c);
        out[nt] = c;
    }
}

__global__ __launch_bounds__(256) void flow_kernel(
    const float* __restrict__ x, const int* __restrict__ perms,
    const _Float16* __restrict__ nvpW0t, const _Float16* __restrict__ nvpW1t, const _Float16* __restrict__ nvpW2t,
    const _Float16* __restrict__ odeW0t, const _Float16* __restrict__ odeW1t, const _Float16* __restrict__ odeW2t,
    const float* __restrict__ biasWs,
    const float* __restrict__ bn1m, const float* __restrict__ bn1v,
    const float* __restrict__ bn1g, const float* __restrict__ bn1b,
    const float* __restrict__ bn2m, const float* __restrict__ bn2v,
    const float* __restrict__ bn2g, const float* __restrict__ bn2b,
    float* __restrict__ out)
{
    extern __shared__ char smem[];
    _Float16* sWA  = (_Float16*)(smem + OFF_WA);
    _Float16* sWB  = (_Float16*)(smem + OFF_WB);
    _Float16* sWC  = (_Float16*)(smem + OFF_WC);
    float*    sBias= (float*)   (smem + OFF_BIAS);
    float*    sScrA= (float*)   (smem + OFF_SCR);
    _Float16* sH2A = (_Float16*)(smem + OFF_H2);

    const int tid  = threadIdx.x;
    const int wave = tid >> 5, lane = tid & 31;
    const int n16  = lane & 15, sel = lane >> 4;

    float*    scrF = sScrA + wave * 1152;         // 16 x 72 f32 view
    _Float16* scrH = (_Float16*)scrF;             // 16 x 144 f16 view (same bytes)
    _Float16* h2w  = sH2A + wave * (16 * ST_H2);

    const int row0 = blockIdx.x * 128 + wave * 16 + sel * 8;

    // load 16x64 state tile in C-layout registers
    v8f y[4];
#pragma unroll
    for (int nt = 0; nt < 4; ++nt)
#pragma unroll
        for (int r = 0; r < 8; ++r)
            y[nt][r] = x[(row0 + r) * DFEAT + nt * 16 + n16];

    const float dtv = 1.0f / NSTEP, hdt = 0.5f / NSTEP;

#pragma unroll 1
    for (int l = 0; l < NLAYER; ++l) {
        // ======== load NVP weights (f16, transposed) ========
        __syncthreads();
        copy16(sWA, nvpW0t + (size_t)l * 256 * ST_W0_NVP, 256 * ST_W0_NVP * 2, tid);
        copy16(sWB, nvpW1t + (size_t)l * 256 * ST_W1,     256 * ST_W1 * 2,     tid);
        copy16(sWC, nvpW2t + (size_t)l * 64  * ST_W2,     64  * ST_W2 * 2,     tid);
        copy16(sBias, biasWs + (size_t)l * 1408,          576 * 4,             tid);
        __syncthreads();

        // ======== column permutation through f32 scratch ========
        int pc[4];
#pragma unroll
        for (int nt = 0; nt < 4; ++nt) pc[nt] = perms[l * DFEAT + nt * 16 + n16];
#pragma unroll
        for (int nt = 0; nt < 4; ++nt)
#pragma unroll
            for (int r = 0; r < 8; ++r)
                scrF[(sel * 8 + r) * ST_SCR_F + nt * 16 + n16] = y[nt][r];
        wait_lds();
#pragma unroll
        for (int nt = 0; nt < 4; ++nt)
#pragma unroll
            for (int r = 0; r < 8; ++r)
                y[nt][r] = scrF[(sel * 8 + r) * ST_SCR_F + pc[nt]];

        // ======== RealNVP coupling ========
        wait_lds();
        store_tile_f16(scrH, ST_SCR_H, 0,  y[0], lane);   // x0 = cols 0..31
        store_tile_f16(scrH, ST_SCR_H, 16, y[1], lane);
        v8f o[4];
        mlp3<1, false>(scrH, h2w, sWA, ST_W0_NVP, sWB, sWC,
                       sBias + 0, sBias + 256, sBias + 512,
                       0.0f, nullptr, o, lane);
        // shift = o[0],o[1]; log_scale = o[2],o[3]; x1 = y[2],y[3]
#pragma unroll
        for (int r = 0; r < 8; ++r) {
            y[2][r] = y[2][r] * __expf(o[2][r]) + o[0][r];
            y[3][r] = y[3][r] * __expf(o[3][r]) + o[1][r];
        }

        // ======== BN1 (inverse form) ========
#pragma unroll
        for (int nt = 0; nt < 4; ++nt) {
            int c = l * DFEAT + nt * 16 + n16;
            float sc = sqrtf(bn1v[c] + EPSV) / bn1g[c];
            float be = bn1b[c], me = bn1m[c];
            y[nt] = (y[nt] - be) * sc + me;
        }

        // ======== load ODE weights ========
        __syncthreads();
        copy16(sWA, odeW0t + (size_t)l * 256 * ST_W0_ODE, 256 * ST_W0_ODE * 2, tid);
        copy16(sWB, odeW1t + (size_t)l * 256 * ST_W1,     256 * ST_W1 * 2,     tid);
        copy16(sWC, odeW2t + (size_t)l * 64  * ST_W2,     64  * ST_W2 * 2,     tid);
        copy16(sBias, biasWs + (size_t)l * 1408 + 576,    832 * 4,             tid);
        __syncthreads();

        // ======== FFJORD: 8 RK4 steps ========
#pragma unroll 1
        for (int step = 0; step < NSTEP; ++step) {
            float t0 = (float)step * dtv;
            v8f k[4], acc[4];
#pragma unroll
            for (int nt = 0; nt < 4; ++nt)
#pragma unroll
                for (int r = 0; r < 8; ++r) acc[nt][r] = 0.0f;
            wait_lds();
#pragma unroll
            for (int nt = 0; nt < 4; ++nt) store_tile_f16(scrH, ST_SCR_H, nt * 16, y[nt], lane);
#pragma unroll 1
            for (int stage = 0; stage < 4; ++stage) {
                float toff = (stage == 0) ? 0.0f : ((stage == 3) ? dtv : hdt);
                float wsum = (stage == 1 || stage == 2) ? 2.0f : 1.0f;
                mlp3<2, true>(scrH, h2w, sWA, ST_W0_ODE, sWB, sWC,
                              sBias + 0, sBias + 512, sBias + 768,
                              t0 + toff, sBias + 256, k, lane);
#pragma unroll
                for (int nt = 0; nt < 4; ++nt) acc[nt] += wsum * k[nt];
                if (stage < 3) {
                    float cn = (stage == 2) ? dtv : hdt;
                    wait_lds();
#pragma unroll
                    for (int nt = 0; nt < 4; ++nt)
                        store_tile_f16(scrH, ST_SCR_H, nt * 16, y[nt] + cn * k[nt], lane);
                }
            }
#pragma unroll
            for (int nt = 0; nt < 4; ++nt) y[nt] += (dtv / 6.0f) * acc[nt];
        }

        // ======== BN2 ========
#pragma unroll
        for (int nt = 0; nt < 4; ++nt) {
            int c = l * DFEAT + nt * 16 + n16;
            float sc = sqrtf(bn2v[c] + EPSV) / bn2g[c];
            float be = bn2b[c], me = bn2m[c];
            y[nt] = (y[nt] - be) * sc + me;
        }
    }

#pragma unroll
    for (int nt = 0; nt < 4; ++nt)
#pragma unroll
        for (int r = 0; r < 8; ++r)
            out[(row0 + r) * DFEAT + nt * 16 + n16] = y[nt][r];
}

// ---- prep: transpose W[L][Ksrc][N] (f32) -> Wt[L][N][stride] (f16), skipping `skip` top rows ----
__global__ void prep_wt(const float* __restrict__ src, _Float16* __restrict__ dst,
                        int K, int N, int stride, int skip)
{
    int idx = blockIdx.x * blockDim.x + threadIdx.x;
    int total = NLAYER * N * K;
    if (idx >= total) return;
    int l = idx / (N * K); int rem = idx % (N * K);
    int n = rem / K;       int k = rem % K;
    dst[((size_t)l * N + n) * stride + k] =
        (_Float16)src[((size_t)l * (K + skip) + skip + k) * N + n];
}

// ---- prep: pack per-layer bias block of 1408 floats ----
__global__ void prep_bias(const float* __restrict__ nvpb0, const float* __restrict__ nvpb1,
                          const float* __restrict__ nvpb2, const float* __restrict__ odeW0,
                          const float* __restrict__ odeb0, const float* __restrict__ odeb1,
                          const float* __restrict__ odeb2, float* __restrict__ dst)
{
    int idx = blockIdx.x * blockDim.x + threadIdx.x;
    if (idx >= NLAYER * 1408) return;
    int l = idx / 1408, j = idx % 1408;
    float v;
    if      (j < 256)  v = nvpb0[l * 256 + j];
    else if (j < 512)  v = nvpb1[l * 256 + (j - 256)];
    else if (j < 576)  v = nvpb2[l * 64  + (j - 512)];
    else if (j < 832)  v = odeb0[l * 256 + (j - 576)];
    else if (j < 1088) v = odeW0[(size_t)l * 65 * 256 + (j - 832)];   // row 0 of ode_W0 = t-row
    else if (j < 1344) v = odeb1[l * 256 + (j - 1088)];
    else               v = odeb2[l * 64  + (j - 1344)];
    dst[idx] = v;
}

extern "C" void kernel_launch(void* const* d_in, const int* in_sizes, int n_in,
                              void* d_out, int out_size, void* d_ws, size_t ws_size,
                              hipStream_t stream)
{
    (void)in_sizes; (void)n_in; (void)out_size; (void)ws_size;
    const float* x      = (const float*)d_in[0];
    const int*   perms  = (const int*)  d_in[1];
    const float* nvp_W0 = (const float*)d_in[2];
    const float* nvp_b0 = (const float*)d_in[3];
    const float* nvp_W1 = (const float*)d_in[4];
    const float* nvp_b1 = (const float*)d_in[5];
    const float* nvp_W2 = (const float*)d_in[6];
    const float* nvp_b2 = (const float*)d_in[7];
    const float* ode_W0 = (const float*)d_in[8];
    const float* ode_b0 = (const float*)d_in[9];
    const float* ode_W1 = (const float*)d_in[10];
    const float* ode_b1 = (const float*)d_in[11];
    const float* ode_W2 = (const float*)d_in[12];
    const float* ode_b2 = (const float*)d_in[13];
    const float* bn1m = (const float*)d_in[14];
    const float* bn1v = (const float*)d_in[15];
    const float* bn1g = (const float*)d_in[16];
    const float* bn1b = (const float*)d_in[17];
    const float* bn2m = (const float*)d_in[18];
    const float* bn2v = (const float*)d_in[19];
    const float* bn2g = (const float*)d_in[20];
    const float* bn2b = (const float*)d_in[21];

    char* ws = (char*)d_ws;
    size_t off = 0;
    auto alloc = [&](size_t bytes) { char* p = ws + off; off += (bytes + 255) & ~(size_t)255; return p; };
    _Float16* nvpW0t = (_Float16*)alloc((size_t)NLAYER * 256 * ST_W0_NVP * 2);
    _Float16* nvpW1t = (_Float16*)alloc((size_t)NLAYER * 256 * ST_W1 * 2);
    _Float16* nvpW2t = (_Float16*)alloc((size_t)NLAYER * 64  * ST_W2 * 2);
    _Float16* odeW0t = (_Float16*)alloc((size_t)NLAYER * 256 * ST_W0_ODE * 2);
    _Float16* odeW1t = (_Float16*)alloc((size_t)NLAYER * 256 * ST_W1 * 2);
    _Float16* odeW2t = (_Float16*)alloc((size_t)NLAYER * 64  * ST_W2 * 2);
    float*    biasWs = (float*)   alloc((size_t)NLAYER * 1408 * 4);

    auto launch_wt = [&](const float* src, _Float16* dst, int K, int N, int stride, int skip) {
        int total = NLAYER * N * K;
        prep_wt<<<(total + 255) / 256, 256, 0, stream>>>(src, dst, K, N, stride, skip);
    };
    launch_wt(nvp_W0, nvpW0t, 32,  256, ST_W0_NVP, 0);
    launch_wt(nvp_W1, nvpW1t, 256, 256, ST_W1,     0);
    launch_wt(nvp_W2, nvpW2t, 256, 64,  ST_W2,     0);
    launch_wt(ode_W0, odeW0t, 64,  256, ST_W0_ODE, 1);   // skip t-row
    launch_wt(ode_W1, odeW1t, 256, 256, ST_W1,     0);
    launch_wt(ode_W2, odeW2t, 256, 64,  ST_W2,     0);
    prep_bias<<<(NLAYER * 1408 + 255) / 256, 256, 0, stream>>>(
        nvp_b0, nvp_b1, nvp_b2, ode_W0, ode_b0, ode_b1, ode_b2, biasWs);

    flow_kernel<<<BATCH / 128, 256, LDS_TOTAL, stream>>>(
        x, perms, nvpW0t, nvpW1t, nvpW2t, odeW0t, odeW1t, odeW2t, biasWs,
        bn1m, bn1v, bn1g, bn1b, bn2m, bn2v, bn2g, bn2b, (float*)d_out);
}
